// ConditionalRQSplineCouplingLayer_38886633898112
// MI455X (gfx1250) — matrix operand order repaired and checked by reading, not compiled, not run on hardware
//
#include <hip/hip_runtime.h>

// ---------------------------------------------------------------------------
// Fused conditional RQ-spline coupling layer for gfx1250 (MI455X).
//   params = relu(relu([z1|cond] @ W1 + b1) @ W2 + b2) @ W3 + b3  (WMMA bf16)
//   spline + logdet fused in-kernel; activations live in LDS only.
// ---------------------------------------------------------------------------

typedef __attribute__((ext_vector_type(16))) __bf16 v16bf;
typedef __attribute__((ext_vector_type(8)))  float  v8f;

#define BATCH_   131072
#define MTILE    64          // rows per workgroup
#define HID      512
#define NPAD     1024        // 32 dims * 32 padded cols (25 real + 7 pad)

// ---- weight repack kernels (f32 row-major -> bf16 col-major [N][K]) -------
__global__ void repack_w1(const float* __restrict__ W1, __bf16* __restrict__ W1t) {
    int i = blockIdx.x * 256 + threadIdx.x;
    if (i < 512 * 96) {
        int n = i / 96, k = i % 96;
        W1t[i] = (__bf16)W1[k * 512 + n];
    }
}
__global__ void repack_w2(const float* __restrict__ W2, __bf16* __restrict__ W2t) {
    int i = blockIdx.x * 256 + threadIdx.x;
    if (i < 512 * 512) {
        int n = i / 512, k = i % 512;
        W2t[i] = (__bf16)W2[k * 512 + n];
    }
}
__global__ void repack_w3(const float* __restrict__ W3, __bf16* __restrict__ W3t) {
    int i = blockIdx.x * 256 + threadIdx.x;
    if (i < NPAD * 512) {
        int n = i / 512, k = i % 512;
        int dim = n >> 5, c = n & 31;
        float v = (c < 25) ? W3[k * 800 + dim * 25 + c] : 0.0f;
        W3t[i] = (__bf16)v;
    }
}
__global__ void repack_b3(const float* __restrict__ b3, float* __restrict__ b3p) {
    int n = blockIdx.x * 256 + threadIdx.x;
    if (n < NPAD) {
        int dim = n >> 5, c = n & 31;
        b3p[n] = (c < 25) ? b3[dim * 25 + c] : 0.0f;
    }
}

// ---- WMMA fragment loaders ------------------------------------------------
// A frag (16x32 bf16, M x K): lane&15 = M row; lane>=16 shifts K groups by 8.
__device__ __forceinline__ v16bf a_frag(const __bf16* H, int ld, int row0,
                                        int kk, int lane) {
    int r  = row0 + (lane & 15);
    int kb = kk + ((lane >> 4) << 3);
    const __bf16* p = H + r * ld + kb;
    v16bf a;
#pragma unroll
    for (int j = 0; j < 8; ++j) a[j] = p[j];
#pragma unroll
    for (int j = 0; j < 8; ++j) a[8 + j] = p[16 + j];
    return a;
}
// B frag (32x16 bf16, K x N) from transposed weights Wt[N][Kp]:
// lane&15 = N col; lane>=16 shifts K groups by 8 (same K order as a_frag).
__device__ __forceinline__ v16bf b_frag(const __bf16* Wt, int Kp, int col,
                                        int kk, int lane) {
    const __bf16* p = Wt + col * Kp + kk + ((lane >> 4) << 3);
    v16bf b;
#pragma unroll
    for (int j = 0; j < 8; ++j) { b[j] = p[j]; b[8 + j] = p[16 + j]; }
    return b;
}

#define WMMA_BF16(A, B, C) \
    __builtin_amdgcn_wmma_f32_16x16x32_bf16(false, (A), false, (B), (short)0, (C), false, false)

// ---- fused kernel ---------------------------------------------------------
__global__ __launch_bounds__(256) void rqspline_fused(
    const float* __restrict__ z, const float* __restrict__ cond,
    const float* __restrict__ b1, const float* __restrict__ b2,
    const __bf16* __restrict__ W1t, const __bf16* __restrict__ W2t,
    const __bf16* __restrict__ W3t, const float* __restrict__ b3p,
    float* __restrict__ out, int batch) {

    __shared__ __align__(16) __bf16 xs [MTILE * 96];   // 12 KB
    __shared__ __align__(16) __bf16 h1s[MTILE * HID];  // 64 KB (reused as f32 scratch in stage 3)
    __shared__ __align__(16) __bf16 h2s[MTILE * HID];  // 64 KB
    __shared__ float logdet_s[MTILE];

    const int tid  = threadIdx.x;
    const int lane = tid & 31;
    const int wid  = tid >> 5;
    const int row0g = blockIdx.x * MTILE;

    if (tid < MTILE) logdet_s[tid] = 0.0f;

    // stage 0: build x = [z1 | cond] in bf16, pass z1 through to output
    for (int i = tid; i < MTILE * 96; i += 256) {
        int r = i / 96, c = i % 96;
        float v = (c < 32) ? z[(row0g + r) * 64 + c]
                           : cond[(row0g + r) * 64 + (c - 32)];
        xs[i] = (__bf16)v;
    }
    for (int i = tid; i < MTILE * 32; i += 256) {
        int r = i / 32, c = i % 32;
        out[(row0g + r) * 64 + c] = z[(row0g + r) * 64 + c];
    }
    __syncthreads();

    // ---- stage 1: h1 = relu(x @ W1 + b1), K=96 ----------------------------
    for (int ct = 0; ct < 4; ++ct) {
        int n0 = wid * 64 + ct * 16;
        v8f acc[4];
#pragma unroll
        for (int rt = 0; rt < 4; ++rt) acc[rt] = (v8f){};
        for (int kk = 0; kk < 96; kk += 32) {
            v16bf b = b_frag(W1t, 96, n0 + (lane & 15), kk, lane);
#pragma unroll
            for (int rt = 0; rt < 4; ++rt) {
                v16bf a = a_frag(xs, 96, rt * 16, kk, lane);
                acc[rt] = WMMA_BF16(a, b, acc[rt]);
            }
        }
        int c = n0 + (lane & 15);
        float bias = b1[c];
        int rb = (lane >> 4) << 3;
#pragma unroll
        for (int rt = 0; rt < 4; ++rt)
#pragma unroll
            for (int j = 0; j < 8; ++j) {
                float v = acc[rt][j] + bias;
                h1s[(rt * 16 + rb + j) * HID + c] = (__bf16)(v > 0.0f ? v : 0.0f);
            }
    }
    __syncthreads();

    // ---- stage 2: h2 = relu(h1 @ W2 + b2), K=512 --------------------------
    for (int ct = 0; ct < 4; ++ct) {
        int n0 = wid * 64 + ct * 16;
        v8f acc[4];
#pragma unroll
        for (int rt = 0; rt < 4; ++rt) acc[rt] = (v8f){};
        for (int kk = 0; kk < HID; kk += 32) {
            v16bf b = b_frag(W2t, HID, n0 + (lane & 15), kk, lane);
#pragma unroll
            for (int rt = 0; rt < 4; ++rt) {
                v16bf a = a_frag(h1s, HID, rt * 16, kk, lane);
                acc[rt] = WMMA_BF16(a, b, acc[rt]);
            }
        }
        int c = n0 + (lane & 15);
        float bias = b2[c];
        int rb = (lane >> 4) << 3;
#pragma unroll
        for (int rt = 0; rt < 4; ++rt)
#pragma unroll
            for (int j = 0; j < 8; ++j) {
                float v = acc[rt][j] + bias;
                h2s[(rt * 16 + rb + j) * HID + c] = (__bf16)(v > 0.0f ? v : 0.0f);
            }
    }
    __syncthreads();

    // ---- stage 3: params = h2 @ W3 + b3, fused spline ---------------------
    // h1s region reused as per-wave 16x32 f32 scratch (2 KB each).
    float* sw = (float*)h1s + wid * (16 * 32);

    for (int dd = 0; dd < 4; ++dd) {
        int dim = wid * 4 + dd;                 // spline dimension 0..31
        for (int rt = 0; rt < 4; ++rt) {
            v8f acc0 = (v8f){}, acc1 = (v8f){};
            for (int kk = 0; kk < HID; kk += 32) {
                v16bf a  = a_frag(h2s, HID, rt * 16, kk, lane);
                v16bf bb0 = b_frag(W3t, HID, dim * 32 + (lane & 15), kk, lane);
                v16bf bb1 = b_frag(W3t, HID, dim * 32 + 16 + (lane & 15), kk, lane);
                acc0 = WMMA_BF16(a, bb0, acc0);
                acc1 = WMMA_BF16(a, bb1, acc1);
            }
            // dump 16x32 tile (rows x param-cols) into wave-private scratch
            int c = lane & 15;
            int rb = (lane >> 4) << 3;
#pragma unroll
            for (int j = 0; j < 8; ++j) {
                sw[(rb + j) * 32 + c]      = acc0[j] + b3p[dim * 32 + c];
                sw[(rb + j) * 32 + 16 + c] = acc1[j] + b3p[dim * 32 + 16 + c];
            }
            // same-wave LDS ops are in-order: safe to read back immediately.
            if (lane < 16) {
                const float* p = sw + lane * 32;  // 25 params for this row/dim
                int   rloc = rt * 16 + lane;
                int   rg   = row0g + rloc;
                float xin  = z[rg * 64 + 32 + dim];

                float pw[8], ph[8], pd[9];
#pragma unroll
                for (int i = 0; i < 8; ++i) { pw[i] = p[i]; ph[i] = p[8 + i]; }
#pragma unroll
                for (int i = 0; i < 9; ++i) pd[i] = p[16 + i];

                // softmax widths -> cumwidths
                float mw = pw[0];
#pragma unroll
                for (int i = 1; i < 8; ++i) mw = fmaxf(mw, pw[i]);
                float ew[8], sumw = 0.0f;
#pragma unroll
                for (int i = 0; i < 8; ++i) { ew[i] = __expf(pw[i] - mw); sumw += ew[i]; }
                float cumw[9]; cumw[0] = -3.0f;
                float invw = 1.0f / sumw;
#pragma unroll
                for (int i = 0; i < 8; ++i)
                    cumw[i + 1] = cumw[i] + (ew[i] * invw * (6.0f - 0.008f) + 0.001f);

                // softmax heights -> cumheights
                float mh = ph[0];
#pragma unroll
                for (int i = 1; i < 8; ++i) mh = fmaxf(mh, ph[i]);
                float eh[8], sumh = 0.0f;
#pragma unroll
                for (int i = 0; i < 8; ++i) { eh[i] = __expf(ph[i] - mh); sumh += eh[i]; }
                float cumh[9]; cumh[0] = -3.0f;
                float invh = 1.0f / sumh;
#pragma unroll
                for (int i = 0; i < 8; ++i)
                    cumh[i + 1] = cumh[i] + (eh[i] * invh * (6.0f - 0.008f) + 0.001f);

                // softplus derivatives
#pragma unroll
                for (int i = 0; i < 9; ++i)
                    pd[i] = fminf(log1pf(__expf(pd[i])) + 0.001f, 10.0f);

                bool  inside = (xin >= -3.0f) && (xin <= 3.0f);
                float xc = fminf(fmaxf(xin, -3.0f), 3.0f);
                int   bi = 0;
#pragma unroll
                for (int k = 1; k <= 8; ++k) bi += (xc >= cumw[k]) ? 1 : 0;
                if (bi > 7) bi = 7;

                float in_cumw  = cumw[bi];
                float in_bw    = fmaxf(cumw[bi + 1] - cumw[bi], 1e-6f);
                float out_cumh = cumh[bi];
                float out_bh   = cumh[bi + 1] - cumh[bi];
                float delta    = out_bh / in_bw;
                float dl = pd[bi], dr = pd[bi + 1];
                float theta = (xc - in_cumw) / in_bw;
                float tomt  = theta * (1.0f - theta);
                float numer = out_bh * (delta * theta * theta + dl * tomt);
                float denom = fmaxf(delta + (dl + dr - 2.0f * delta) * tomt, 1e-6f);
                float oin   = out_cumh + numer / denom;
                float dnum  = delta * delta *
                    (dr * theta * theta + 2.0f * delta * tomt +
                     dl * (1.0f - theta) * (1.0f - theta));
                float dden  = denom * denom;
                dnum = fmaxf(dnum, 1e-6f);
                dden = fmaxf(dden, 1e-6f);
                float lad = __logf(dnum) - __logf(dden) - __logf(in_bw);

                float o    = inside ? oin : xin;
                float ladf = inside ? lad : 0.0f;
                if (__builtin_isnan(o))       o = 0.0f;
                else if (__builtin_isinf(o))  o = (o > 0.0f) ? 3.0f : -3.0f;
                if (!__builtin_isfinite(ladf)) ladf = 0.0f;

                out[rg * 64 + 32 + dim] = o;
                atomicAdd(&logdet_s[rloc], ladf);
            }
        }
    }
    __syncthreads();
    if (tid < MTILE)
        out[(size_t)batch * 64 + row0g + tid] = logdet_s[tid];
}

// ---- launch ---------------------------------------------------------------
extern "C" void kernel_launch(void* const* d_in, const int* in_sizes, int n_in,
                              void* d_out, int out_size, void* d_ws, size_t ws_size,
                              hipStream_t stream) {
    const float* z    = (const float*)d_in[0];
    const float* cond = (const float*)d_in[1];
    const float* W1   = (const float*)d_in[2];
    const float* b1   = (const float*)d_in[3];
    const float* W2   = (const float*)d_in[4];
    const float* b2   = (const float*)d_in[5];
    const float* W3   = (const float*)d_in[6];
    const float* b3   = (const float*)d_in[7];
    float* out = (float*)d_out;

    char* ws = (char*)d_ws;
    __bf16* W1t = (__bf16*)(ws);
    __bf16* W2t = (__bf16*)(ws + 98304);
    __bf16* W3t = (__bf16*)(ws + 98304 + 524288);
    float*  b3p = (float*) (ws + 98304 + 524288 + 1048576);

    int batch = in_sizes[0] / 64;

    repack_w1<<<(512 * 96  + 255) / 256, 256, 0, stream>>>(W1, W1t);
    repack_w2<<<(512 * 512 + 255) / 256, 256, 0, stream>>>(W2, W2t);
    repack_w3<<<(NPAD * 512 + 255) / 256, 256, 0, stream>>>(W3, W3t);
    repack_b3<<<(NPAD + 255) / 256, 256, 0, stream>>>(b3, b3p);

    rqspline_fused<<<batch / MTILE, 256, 0, stream>>>(
        z, cond, b1, b2, W1t, W2t, W3t, b3p, out, batch);
}
